// AJRNN_45492293599888
// MI455X (gfx1250) — matrix operand order repaired and checked by reading, not compiled, not run on hardware
//
#include <hip/hip_runtime.h>
#include <hip/hip_bf16.h>

// ---------------------------------------------------------------------------
// AJRNN (2-layer LSTM + imputation projection) for MI455X / gfx1250.
// B=128, T=256, D=64, H=512.  One workgroup per 16-row batch tile (8 WGs),
// 16 waves per WG, persistent over all 256 timesteps.  bf16 WMMA
// (v_wmma_f32_16x16x32_bf16) with f32 accumulation; weights converted and
// transposed to N-major bf16 once into d_ws (L2-resident, ~6.6 MB); h state
// double-buffered in LDS; c state in per-wave VGPRs.
//
// Round-2 fixes:
//  * Weight loads now go through explicit address_space(1) row pointers ->
//    global_load_b128 (round 2 emitted flat_load_b128, which ties up DScnt
//    and the LDS path alongside the real LDS A-operand traffic).
//  * One laundered row pointer per (gate) per phase; K-chunks indexed with
//    immediate offsets instead of per-load 64-bit address math.
// ---------------------------------------------------------------------------

typedef __bf16 bf16_t;
typedef __attribute__((ext_vector_type(16))) __bf16 v16bf;
typedef __attribute__((ext_vector_type(8)))  float  v8f;
typedef __attribute__((ext_vector_type(4)))  unsigned int u32x4;

// global-address-space row pointer for B operands (16-byte granules)
typedef const __attribute__((address_space(1))) u32x4* gbp;

#define BB   128
#define TT   256
#define DD   64
#define HH   512
#define OUTC ((TT - 1) * DD)   // 16320

// total bf16 elements of converted weights in d_ws
#define WS_ELEMS ((size_t)2048 * 64 + 3 * (size_t)2048 * 512 + (size_t)64 * 512)

union BF32B { v16bf v; u32x4 u[2]; };

__device__ __forceinline__ v8f v8f_zero() {
  v8f z = {0.f, 0.f, 0.f, 0.f, 0.f, 0.f, 0.f, 0.f};
  return z;
}

// A operand (16x32 bf16) from row-major LDS tile, row stride `strideElems`.
// Lane layout (this code's contiguous k order, not the ISA table order): m = lane&15, khalf = lane>>4;
// VGPR0-7 = K[khalf*16 .. +15] (contiguous, matches B operand).
__device__ __forceinline__ v16bf load_A_lds(const bf16_t* base, int strideElems, int lane) {
  const int m  = lane & 15;
  const int kh = lane >> 4;
  const bf16_t* p = base + m * strideElems + kh * 16;
  BF32B r;
  r.u[0] = *(const u32x4*)(p);
  r.u[1] = *(const u32x4*)(p + 8);
  return r.v;
}

// Build a laundered global-AS pointer to this lane's slice of weight row
// block nRow..nRow+15 (N-major weights wt[N][K]).  Lane layout: n = lane&15,
// half = lane>>4; the lane reads K[half*16 + kc*32 ...] = 32 contiguous bytes
// per K-chunk at byte offset kc*64 from this pointer.
// Laundering blocks LICM across the time loop (prevents the hoist+spill-to-
// scratch the compiler did in round 1) while keeping immediate-offset loads.
__device__ __forceinline__ gbp make_B_row(const bf16_t* wt, int Kelems, int nRow, int lane) {
  const int n    = lane & 15;
  const int half = lane >> 4;
  const bf16_t* p = wt + (size_t)(nRow + n) * Kelems + half * 16;
  gbp gp = (gbp)p;
  asm volatile("" : "+v"(gp));
  return gp;
}

// B operand (32x16 bf16) K-chunk kc: 32 contiguous bytes per lane.
__device__ __forceinline__ v16bf load_B_at(gbp gp, int kc) {
  BF32B r;
  r.u[0] = gp[kc * 4 + 0];   // global_load_b128, ioffset = kc*64
  r.u[1] = gp[kc * 4 + 1];   // global_load_b128, ioffset = kc*64+16
  return r.v;
}

__device__ __forceinline__ v8f wmma_bf16(v16bf a, v16bf b, v8f c) {
  return __builtin_amdgcn_wmma_f32_16x16x32_bf16(
      /*neg_a=*/false, a, /*neg_b=*/false, b,
      /*c_mod=*/(short)0, c, /*reuse_a=*/false, /*reuse_b=*/false);
}

__device__ __forceinline__ float sigmoidf_fast(float x) {
  return 1.0f / (1.0f + __expf(-x));
}

// ---------------------------------------------------------------------------
// Prep: fp32 [K][N] row-major  ->  bf16 [N][K] row-major (transpose+convert).
// ---------------------------------------------------------------------------
__global__ void convert_transpose_kernel(const float* __restrict__ src,
                                         bf16_t* __restrict__ dst, int K, int N) {
  const size_t total = (size_t)K * (size_t)N;
  for (size_t i = (size_t)blockIdx.x * blockDim.x + threadIdx.x; i < total;
       i += (size_t)gridDim.x * blockDim.x) {
    const size_t n = i / (size_t)K;
    const size_t k = i % (size_t)K;
    dst[i] = (bf16_t)src[k * (size_t)N + n];
  }
}

// ---------------------------------------------------------------------------
// Main persistent kernel: one block per 16-row batch tile, 16 waves.
// ---------------------------------------------------------------------------
__global__ __launch_bounds__(512, 1)
void ajrnn_main(const float*  __restrict__ x,       // [B,T,D]
                const float*  __restrict__ biasD,   // [D]
                const float*  __restrict__ bias1,   // [4H]
                const float*  __restrict__ bias2,   // [4H]
                const bf16_t* __restrict__ k1t,     // [4H][D]  (N-major)
                const bf16_t* __restrict__ r1t,     // [4H][H]
                const bf16_t* __restrict__ k2t,     // [4H][H]
                const bf16_t* __restrict__ r2t,     // [4H][H]
                const bf16_t* __restrict__ Wt,      // [D][H]
                float* __restrict__ out)            // [B, (T-1)*D]
{
  __shared__ bf16_t h1s[2][16][HH];   // 32 KB, ping-pong
  __shared__ bf16_t h2s[2][16][HH];   // 32 KB, ping-pong
  __shared__ bf16_t curs[16][DD];     //  2 KB, current (possibly imputed) input

  const int m0    = blockIdx.x * 16;
  const int tid   = threadIdx.x;
  const int wave  = tid >> 5;         // 0..15: owns h-columns [wave*32, wave*32+32)
  const int lane  = tid & 31;
  const int ln    = lane & 15;
  const int mBase = (lane >> 4) * 8;  // C/D row base for this lane

  // One-time grid-strided L2 warm of the whole bf16 weight image (k1t is the
  // base of d_ws): ~6.6 MB / 4096 threads = ~13 cachelines per thread.
  {
    const char* wsBase = (const char*)k1t;
    const size_t wsBytes = WS_ELEMS * sizeof(bf16_t);
    const size_t gtid = (size_t)blockIdx.x * 512 + tid;
    for (size_t off = gtid * 128; off < wsBytes; off += (size_t)8 * 512 * 128)
      __builtin_prefetch(wsBase + off, 0, 0);
  }

  // h(0) = 0 into read buffer 0; cur(0) = x[:,0,:] verbatim (t==0: no imputation).
  for (int i = tid; i < 16 * HH; i += 512) {
    (&h1s[0][0][0])[i] = (bf16_t)0.0f;
    (&h2s[0][0][0])[i] = (bf16_t)0.0f;
  }
  for (int i = tid; i < 16 * DD; i += 512) {
    const int m = i >> 6, d = i & 63;
    (&curs[0][0])[i] = (bf16_t)x[(size_t)(m0 + m) * (TT * DD) + d];
  }

  // Bias preload (accumulator layout: column = gate*H + wave*32 + jb*16 + ln).
  float b1r[2][4], b2r[2][4];
#pragma unroll
  for (int jb = 0; jb < 2; ++jb)
#pragma unroll
    for (int g = 0; g < 4; ++g) {
      const int col = g * HH + wave * 32 + jb * 16 + ln;
      b1r[jb][g] = bias1[col];
      b2r[jb][g] = bias2[col];
    }
  const float bpr = (wave < 4) ? biasD[wave * 16 + ln] : 0.0f;

  // Cell state in registers (wave-private column slices).
  v8f c1[2] = { v8f_zero(), v8f_zero() };
  v8f c2[2] = { v8f_zero(), v8f_zero() };

  __syncthreads();

  for (int t = 0; t < TT; ++t) {
    const int rd = t & 1, wr = rd ^ 1;

    // ---------------- Layer 1: z1 = cur@K1 + h1_old@R1 + b1 ----------------
#pragma unroll
    for (int jb = 0; jb < 2; ++jb) {
      v8f acc[4] = { v8f_zero(), v8f_zero(), v8f_zero(), v8f_zero() };
      const int nrb0 = wave * 32 + jb * 16;

      gbp bk[4], br[4];
#pragma unroll
      for (int g = 0; g < 4; ++g) {
        bk[g] = make_B_row(k1t, DD, g * HH + nrb0, lane);
        br[g] = make_B_row(r1t, HH, g * HH + nrb0, lane);
      }

#pragma unroll
      for (int kc = 0; kc < 2; ++kc) {           // input term, K = 64
        const v16bf a = load_A_lds(&curs[0][0] + kc * 32, DD, lane);
#pragma unroll
        for (int g = 0; g < 4; ++g)
          acc[g] = wmma_bf16(a, load_B_at(bk[g], kc), acc[g]);
      }
#pragma unroll 2
      for (int kc = 0; kc < 16; ++kc) {          // recurrent term, K = 512
        const v16bf a = load_A_lds(&h1s[rd][0][0] + kc * 32, HH, lane);
#pragma unroll
        for (int g = 0; g < 4; ++g)
          acc[g] = wmma_bf16(a, load_B_at(br[g], kc), acc[g]);
      }
      // Fused gates (i,f,g,o) -> c1, h1_new.
      const int col = wave * 32 + jb * 16 + ln;
#pragma unroll
      for (int r = 0; r < 8; ++r) {
        const float ii = sigmoidf_fast(acc[0][r] + b1r[jb][0]);
        const float ff = sigmoidf_fast(acc[1][r] + b1r[jb][1]);
        const float gg = tanhf(acc[2][r] + b1r[jb][2]);
        const float oo = sigmoidf_fast(acc[3][r] + b1r[jb][3]);
        const float cn = ff * c1[jb][r] + ii * gg;
        c1[jb][r] = cn;
        h1s[wr][mBase + r][col] = (bf16_t)(oo * tanhf(cn));
      }
    }
    __syncthreads();

    // ---------------- Layer 2: z2 = h1_new@K2 + h2_old@R2 + b2 --------------
#pragma unroll
    for (int jb = 0; jb < 2; ++jb) {
      v8f acc[4] = { v8f_zero(), v8f_zero(), v8f_zero(), v8f_zero() };
      const int nrb0 = wave * 32 + jb * 16;

      gbp bk[4], br[4];
#pragma unroll
      for (int g = 0; g < 4; ++g) {
        bk[g] = make_B_row(k2t, HH, g * HH + nrb0, lane);
        br[g] = make_B_row(r2t, HH, g * HH + nrb0, lane);
      }

#pragma unroll 2
      for (int kc = 0; kc < 16; ++kc) {          // feed-forward term, K = 512
        const v16bf a = load_A_lds(&h1s[wr][0][0] + kc * 32, HH, lane);
#pragma unroll
        for (int g = 0; g < 4; ++g)
          acc[g] = wmma_bf16(a, load_B_at(bk[g], kc), acc[g]);
      }
#pragma unroll 2
      for (int kc = 0; kc < 16; ++kc) {          // recurrent term, K = 512
        const v16bf a = load_A_lds(&h2s[rd][0][0] + kc * 32, HH, lane);
#pragma unroll
        for (int g = 0; g < 4; ++g)
          acc[g] = wmma_bf16(a, load_B_at(br[g], kc), acc[g]);
      }
      const int col = wave * 32 + jb * 16 + ln;
#pragma unroll
      for (int r = 0; r < 8; ++r) {
        const float ii = sigmoidf_fast(acc[0][r] + b2r[jb][0]);
        const float ff = sigmoidf_fast(acc[1][r] + b2r[jb][1]);
        const float gg = tanhf(acc[2][r] + b2r[jb][2]);
        const float oo = sigmoidf_fast(acc[3][r] + b2r[jb][3]);
        const float cn = ff * c2[jb][r] + ii * gg;
        c2[jb][r] = cn;
        h2s[wr][mBase + r][col] = (bf16_t)(oo * tanhf(cn));
      }
    }
    __syncthreads();

    // ---- pred = h2_new @ W^T + biasD : output row t, and imputed input t+1 ----
    if (t < TT - 1) {
      if (wave < 4) {                            // 4 N-tiles of 16 (D = 64)
        v8f acc = v8f_zero();
        gbp bw = make_B_row(Wt, HH, wave * 16, lane);
#pragma unroll 2
        for (int kc = 0; kc < 16; ++kc) {        // K = 512
          const v16bf a = load_A_lds(&h2s[wr][0][0] + kc * 32, HH, lane);
          acc = wmma_bf16(a, load_B_at(bw, kc), acc);
        }
        const int d = wave * 16 + ln;
#pragma unroll
        for (int r = 0; r < 8; ++r) {
          const int   m  = mBase + r;
          const float p  = acc[r] + bpr;
          out[(size_t)(m0 + m) * OUTC + (size_t)t * DD + d] = p;
          const float xv = x[(size_t)(m0 + m) * (TT * DD) + (size_t)(t + 1) * DD + d];
          curs[m][d] = (bf16_t)((xv == 128.0f) ? p : xv);   // MISSING sentinel
        }
      }
    }
    __syncthreads();
  }
}

// ---------------------------------------------------------------------------
// Launcher.  Inputs (setup_inputs order):
//  0 input[B,T,D] 1 W[H,D] 2 bias[D] 3 kernel1[D,4H] 4 rec1[H,4H] 5 bias1[4H]
//  6 kernel2[H,4H] 7 rec2[H,4H] 8 bias2[4H].  Output fp32 [B,(T-1)*D].
// d_ws holds the bf16 N-major weight copies (~6.6 MB), rebuilt every call.
// ---------------------------------------------------------------------------
extern "C" void kernel_launch(void* const* d_in, const int* in_sizes, int n_in,
                              void* d_out, int out_size, void* d_ws, size_t ws_size,
                              hipStream_t stream) {
  const float* x     = (const float*)d_in[0];
  const float* Wm    = (const float*)d_in[1];
  const float* biasD = (const float*)d_in[2];
  const float* k1    = (const float*)d_in[3];
  const float* r1    = (const float*)d_in[4];
  const float* b1    = (const float*)d_in[5];
  const float* k2    = (const float*)d_in[6];
  const float* r2    = (const float*)d_in[7];
  const float* b2    = (const float*)d_in[8];
  float* out = (float*)d_out;

  bf16_t* ws  = (bf16_t*)d_ws;
  bf16_t* k1t = ws;                            // [2048][64]
  bf16_t* r1t = k1t + (size_t)2048 * 64;       // [2048][512]
  bf16_t* k2t = r1t + (size_t)2048 * 512;      // [2048][512]
  bf16_t* r2t = k2t + (size_t)2048 * 512;      // [2048][512]
  bf16_t* Wt  = r2t + (size_t)2048 * 512;      // [64][512]

  convert_transpose_kernel<<<256, 256, 0, stream>>>(k1, k1t, 64,  2048);
  convert_transpose_kernel<<<512, 256, 0, stream>>>(r1, r1t, 512, 2048);
  convert_transpose_kernel<<<512, 256, 0, stream>>>(k2, k2t, 512, 2048);
  convert_transpose_kernel<<<512, 256, 0, stream>>>(r2, r2t, 512, 2048);
  convert_transpose_kernel<<<64,  256, 0, stream>>>(Wm, Wt,  512, 64);

  ajrnn_main<<<8, 512, 0, stream>>>(x, biasD, b1, b2, k1t, r1t, k2t, r2t, Wt, out);
}
